// GraphAttention_13761075216816
// MI455X (gfx1250) — matrix operand order repaired and checked by reading, not compiled, not run on hardware
//
#include <hip/hip_runtime.h>
#include <hip/hip_bf16.h>
#include <math.h>

// Problem constants (match reference)
#define NN   4096
#define FIN  128
#define FF   64
#define HH   8
#define NW32 (NN / 32)   // 128 words of packed adjacency per row

typedef __attribute__((ext_vector_type(16))) _Float16 v16h;
typedef __attribute__((ext_vector_type(8)))  _Float16 v8h;
typedef __attribute__((ext_vector_type(8)))  float    v8f;

__device__ __forceinline__ v8f wmma_f16(v16h a, v16h b, v8f c) {
    // D = A(16x32 f16) * B(32x16 f16) + C(16x16 f32)
    return __builtin_amdgcn_wmma_f32_16x16x32_f16(
        false, a, false, b, (short)0, c, false, false);
}

// ---------------------------------------------------------------------------
// Kernel 0: X (f32) -> Xh (f16), row-major [N, FIN]
// ---------------------------------------------------------------------------
__global__ void k_cvt_x(const float* __restrict__ X, _Float16* __restrict__ Xh) {
    int i = blockIdx.x * blockDim.x + threadIdx.x;
    if (i < NN * FIN) Xh[i] = (_Float16)X[i];
}

// ---------------------------------------------------------------------------
// Kernel 1: pack W [H, FIN, FF] f32 into WMMA B-operand swizzled f16 layout.
// Layout: Wswz[(((h*4 + kc)*4 + t)*512) + lane*16 + klocal]
//   lane 0..31, klocal 0..15 ; K = kc*32 + (klocal&7) + (klocal<8 ? 0 : 16) + (lane<16 ? 0 : 8)
//   f = t*16 + (lane & 15)
// ---------------------------------------------------------------------------
__global__ void k_pack_w(const float* __restrict__ W, _Float16* __restrict__ Wswz) {
    int idx = blockIdx.x * blockDim.x + threadIdx.x;
    if (idx >= HH * FIN * FF) return;
    int klocal = idx & 15;
    int lane   = (idx >> 4) & 31;
    int t      = (idx >> 9) & 3;
    int kc     = (idx >> 11) & 3;
    int h      = idx >> 13;
    int k = kc * 32 + (klocal & 7) + ((klocal < 8) ? 0 : 16) + ((lane < 16) ? 0 : 8);
    int f = t * 16 + (lane & 15);
    Wswz[idx] = (_Float16)W[(h * FIN + k) * FF + f];
}

// ---------------------------------------------------------------------------
// Kernel 2: pack adjacency A [N,N] int32 -> bitmask Apack [N, N/32] u32
// ---------------------------------------------------------------------------
__global__ void k_pack_a(const int* __restrict__ A, unsigned* __restrict__ Apack) {
    int idx = blockIdx.x * blockDim.x + threadIdx.x;
    if (idx >= NN * NW32) return;
    int n = idx >> 7;          // / NW32
    int c = idx & (NW32 - 1);
    const int* row = A + (long)n * NN + c * 32;
    unsigned w = 0;
#pragma unroll
    for (int j = 0; j < 32; ++j) w |= (row[j] != 0 ? 1u : 0u) << j;
    Apack[idx] = w;
}

// ---------------------------------------------------------------------------
// Kernel 3: feats GEMM (per head): feats[h] = Xh @ W[h]  (WMMA f16->f32)
// Each wave: 16 rows x 64 cols. Also computes s_self/s_neigh row dots and
// writes feats into the B-operand-swizzled f16 layout for the attention GEMM.
// feats_swz[(((h*128 + c)*4 + t)*512) + lane*16 + klocal], K = m index.
// grid = (64, H), block = 128 (4 waves)
// ---------------------------------------------------------------------------
__global__ void k_feats(const _Float16* __restrict__ Xh,
                        const _Float16* __restrict__ Wswz,
                        const float* __restrict__ a_self,
                        const float* __restrict__ a_neigh,
                        _Float16* __restrict__ feats_swz,
                        float* __restrict__ s_self,
                        float* __restrict__ s_neigh) {
    const int h    = blockIdx.y;
    const int wave = threadIdx.x >> 5;
    const int l    = threadIdx.x & 31;
    const int rowbase = blockIdx.x * 64 + wave * 16;
    const int arow = rowbase + (l & 15);        // A-fragment row for this lane
    const int koff = (l < 16) ? 0 : 8;          // f16 16x32 A-layout K offset

    v8f acc[4];
#pragma unroll
    for (int t = 0; t < 4; ++t) acc[t] = (v8f)(0.0f);

#pragma unroll
    for (int kc = 0; kc < 4; ++kc) {
        const _Float16* ap = Xh + (long)arow * FIN + kc * 32 + koff;
        v8h a0 = *(const v8h*)ap;
        v8h a1 = *(const v8h*)(ap + 16);
        v16h a;
#pragma unroll
        for (int j = 0; j < 8; ++j) { a[j] = a0[j]; a[8 + j] = a1[j]; }
        const _Float16* wb = Wswz + ((h * 4 + kc) * 4) * 512 + l * 16;
#pragma unroll
        for (int t = 0; t < 4; ++t) {
            v16h b = *(const v16h*)(wb + t * 512);
            acc[t] = wmma_f16(a, b, acc[t]);
        }
    }

    // ---- s_self / s_neigh: per-row dot of feats with a_self/a_neigh ----
    // C layout: acc[t][i] = feats[rowbase + i + (l<16?0:8), t*16 + (l&15)]
    float as[4], an[4];
#pragma unroll
    for (int t = 0; t < 4; ++t) {
        as[t] = a_self [h * FF + t * 16 + (l & 15)];
        an[t] = a_neigh[h * FF + t * 16 + (l & 15)];
    }
    float ps[8], pn[8];
#pragma unroll
    for (int i = 0; i < 8; ++i) {
        float s = 0.f, n = 0.f;
#pragma unroll
        for (int t = 0; t < 4; ++t) { s += acc[t][i] * as[t]; n += acc[t][i] * an[t]; }
        ps[i] = s; pn[i] = n;
    }
#pragma unroll
    for (int off = 8; off >= 1; off >>= 1) {
#pragma unroll
        for (int i = 0; i < 8; ++i) {
            ps[i] += __shfl_xor(ps[i], off, 16);
            pn[i] += __shfl_xor(pn[i], off, 16);
        }
    }
    if ((l & 15) == 0) {
        int r0 = rowbase + ((l < 16) ? 0 : 8);
#pragma unroll
        for (int i = 0; i < 8; ++i) {
            s_self [h * NN + r0 + i] = ps[i];
            s_neigh[h * NN + r0 + i] = pn[i];
        }
    }

    // ---- store feats in B-operand-swizzled f16 layout ----
#pragma unroll
    for (int t = 0; t < 4; ++t) {
#pragma unroll
        for (int i = 0; i < 8; ++i) {
            int m  = rowbase + i + ((l < 16) ? 0 : 8);
            int c  = m >> 5;
            int ko = m & 31;
            int tl = (l & 15) + (((ko >> 3) & 1) ? 16 : 0);
            long idx = ((long)((h * 128 + c) * 4 + t)) * 512 + tl * 16 + ((ko & 7) | ((ko >> 1) & 8));
            feats_swz[idx] = (_Float16)acc[t][i];
        }
    }
}

// ---------------------------------------------------------------------------
// Kernel 4: per-(h,n) softmax statistics (online max & sum over masked row).
// One wave per (h,n). grid = 8192 blocks x 128 threads (4 waves).
// ---------------------------------------------------------------------------
__global__ void k_rowstats(const unsigned* __restrict__ Apack,
                           const float* __restrict__ s_self,
                           const float* __restrict__ s_neigh,
                           float* __restrict__ rowmax,
                           float* __restrict__ rowinv) {
    int gw = blockIdx.x * 4 + (threadIdx.x >> 5);
    int n  = gw & (NN - 1);
    int h  = gw >> 12;
    int l  = threadIdx.x & 31;

    float ss = s_self[h * NN + n];
    const float* sn = s_neigh + h * NN;
    const unsigned* ap = Apack + (long)n * NW32;

    float m = -1e30f, sum = 0.f;
    for (int it = 0; it < NW32; ++it) {
        unsigned word = ap[it];
        float v = ss + sn[it * 32 + l];
        v = v > 0.f ? v : 0.2f * v;             // LeakyReLU(0.2)
        bool bit = (word >> l) & 1u;
        float val = bit ? v : -1e30f;
        float nm = fmaxf(m, val);
        sum = sum * __expf(m - nm) + (bit ? __expf(v - nm) : 0.f);
        m = nm;
    }
#pragma unroll
    for (int off = 16; off >= 1; off >>= 1) {
        float om = __shfl_xor(m, off, 32);
        float os = __shfl_xor(sum, off, 32);
        float nm = fmaxf(m, om);
        sum = sum * __expf(m - nm) + os * __expf(om - nm);
        m = nm;
    }
    if (l == 0) {
        rowmax[h * NN + n] = m;
        rowinv[h * NN + n] = 1.f / sum;
    }
}

// ---------------------------------------------------------------------------
// Kernel 5: fused attention GEMM: out[h] = softmax(scores) @ feats[h].
// Each wave: 16 rows x 64 features, loops over 128 chunks of 32 neighbors.
// Probabilities regenerated on the fly straight into the A fragment (f16).
// grid = (64, H), block = 128 (4 waves).
// ---------------------------------------------------------------------------
__global__ void k_attn(const unsigned* __restrict__ Apack,
                       const float* __restrict__ s_self,
                       const float* __restrict__ s_neigh,
                       const float* __restrict__ rowmax,
                       const float* __restrict__ rowinv,
                       const _Float16* __restrict__ feats_swz,
                       const float* __restrict__ bvec,
                       float* __restrict__ out) {
    const int h    = blockIdx.y;
    const int wave = threadIdx.x >> 5;
    const int l    = threadIdx.x & 31;
    const int rowbase = blockIdx.x * 64 + wave * 16;
    const int row  = rowbase + (l & 15);        // A-fragment row
    const int koff = (l < 16) ? 0 : 8;          // f16 16x32 A-layout K offset

    const float ss = s_self[h * NN + row];
    const float rm = rowmax[h * NN + row];
    const float ri = rowinv[h * NN + row];
    const float* snb = s_neigh + h * NN;
    const unsigned* ap = Apack + (long)row * NW32;
    const _Float16* fsb = feats_swz + ((long)h * 128) * 4 * 512 + l * 16;

    v8f acc[4];
#pragma unroll
    for (int t = 0; t < 4; ++t) acc[t] = (v8f)(0.0f);

    for (int c = 0; c < NW32; ++c) {
        unsigned word = ap[c];
        const float* snp = snb + c * 32 + koff;
        v8f sn0 = *(const v8f*)snp;
        v8f sn1 = *(const v8f*)(snp + 16);
        v16h a;
#pragma unroll
        for (int j = 0; j < 8; ++j) {
            float v = ss + sn0[j];
            v = v > 0.f ? v : 0.2f * v;
            float p = ((word >> (koff + j)) & 1u) ? __expf(v - rm) * ri : 0.f;
            a[j] = (_Float16)p;
        }
#pragma unroll
        for (int j = 0; j < 8; ++j) {
            float v = ss + sn1[j];
            v = v > 0.f ? v : 0.2f * v;
            float p = ((word >> (koff + 16 + j)) & 1u) ? __expf(v - rm) * ri : 0.f;
            a[8 + j] = (_Float16)p;
        }
        const _Float16* fb = fsb + (long)c * 4 * 512;
#pragma unroll
        for (int t = 0; t < 4; ++t) {
            v16h b = *(const v16h*)(fb + t * 512);
            acc[t] = wmma_f16(a, b, acc[t]);
        }
    }

    // Epilogue: bias + ELU + transposed store to [N, H*F]
    float bb[4];
#pragma unroll
    for (int t = 0; t < 4; ++t) bb[t] = bvec[h * FF + t * 16 + (l & 15)];
#pragma unroll
    for (int t = 0; t < 4; ++t) {
#pragma unroll
        for (int i = 0; i < 8; ++i) {
            float x = acc[t][i] + bb[t];
            x = x > 0.f ? x : (__expf(x) - 1.f);          // ELU(alpha=1)
            int ro = rowbase + i + ((l < 16) ? 0 : 8);
            out[(long)ro * (HH * FF) + h * FF + t * 16 + (l & 15)] = x;
        }
    }
}

// ---------------------------------------------------------------------------
// Host launcher
// ---------------------------------------------------------------------------
extern "C" void kernel_launch(void* const* d_in, const int* in_sizes, int n_in,
                              void* d_out, int out_size, void* d_ws, size_t ws_size,
                              hipStream_t stream) {
    const float* X       = (const float*)d_in[0];
    const int*   A       = (const int*)  d_in[1];
    const float* W       = (const float*)d_in[2];
    const float* bvec    = (const float*)d_in[3];
    const float* a_self  = (const float*)d_in[4];
    const float* a_neigh = (const float*)d_in[5];
    float* out = (float*)d_out;

    // Workspace layout (bytes):
    char* ws = (char*)d_ws;
    _Float16* Xh        = (_Float16*)(ws + 0);                 // 1,048,576
    _Float16* Wswz      = (_Float16*)(ws + 1048576);           //   131,072
    unsigned* Apack     = (unsigned*)(ws + 1179648);           // 2,097,152
    float*    s_self    = (float*)   (ws + 3276800);           //   131,072
    float*    s_neigh   = (float*)   (ws + 3407872);           //   131,072
    float*    rowmax    = (float*)   (ws + 3538944);           //   131,072
    float*    rowinv    = (float*)   (ws + 3670016);           //   131,072
    _Float16* feats_swz = (_Float16*)(ws + 3801088);           // 4,194,304  -> ~8 MB total

    k_cvt_x <<<(NN * FIN + 255) / 256, 256, 0, stream>>>(X, Xh);
    k_pack_w<<<(HH * FIN * FF + 255) / 256, 256, 0, stream>>>(W, Wswz);
    k_pack_a<<<(NN * NW32 + 255) / 256, 256, 0, stream>>>(A, Apack);

    dim3 gF(64, HH);
    k_feats<<<gF, 128, 0, stream>>>(Xh, Wswz, a_self, a_neigh,
                                    feats_swz, s_self, s_neigh);

    k_rowstats<<<(HH * NN) / 4, 128, 0, stream>>>(Apack, s_self, s_neigh,
                                                  rowmax, rowinv);

    dim3 gA(64, HH);
    k_attn<<<gA, 128, 0, stream>>>(Apack, s_self, s_neigh, rowmax, rowinv,
                                   feats_swz, bvec, out);
}
